// MultiHeadAttention_53910429499937
// MI455X (gfx1250) — matrix operand order repaired and checked by reading, not compiled, not run on hardware
//
#include <hip/hip_runtime.h>
#include <hip/hip_bf16.h>

// ---------------- types & helpers ----------------
typedef __attribute__((ext_vector_type(8)))  float  v8f;
typedef __attribute__((ext_vector_type(16))) __bf16 v16bf;

struct TwoU4 { uint4 a, b; };

__device__ __forceinline__ unsigned short bfbits(float f) {
  unsigned u = __builtin_bit_cast(unsigned, f);
  u += 0x7FFFu + ((u >> 16) & 1u);           // round-to-nearest-even
  return (unsigned short)(u >> 16);
}
__device__ __forceinline__ __bf16 f2bf(float f) {
  unsigned short h = bfbits(f);
  return __builtin_bit_cast(__bf16, h);
}
__device__ __forceinline__ float bf2f(__bf16 x) {
  unsigned short h = __builtin_bit_cast(unsigned short, x);
  unsigned u = ((unsigned)h) << 16;
  return __builtin_bit_cast(float, u);
}
__device__ __forceinline__ unsigned pk2(float a, float b) {
  return (unsigned)bfbits(a) | ((unsigned)bfbits(b) << 16);
}
// Build a v16bf WMMA fragment from two 16-byte chunks.
__device__ __forceinline__ v16bf mkfrag(const __bf16* p0, const __bf16* p1) {
  TwoU4 u;
  u.a = *(const uint4*)p0;
  u.b = *(const uint4*)p1;
  return __builtin_bit_cast(v16bf, u);
}
__device__ __forceinline__ uint4 pk16(const float4& x, const float4& y) {
  uint4 u;
  u.x = pk2(x.x, x.y); u.y = pk2(x.z, x.w);
  u.z = pk2(y.x, y.y); u.w = pk2(y.z, y.w);
  return u;
}

static constexpr int S_LEN = 2048;
static constexpr int HEADS = 32;
static constexpr int KVH   = 8;
static constexpr int HD    = 128;

// ---------------- GEMM: C[M,N] = A[M,K] * W[N,K]^T (bf16 WMMA, f32 acc) ------
// grid = (N/128, M/128), block = 256 (8 waves). Wave tile 32(M) x 64(N).
// Double-buffered LDS; next tile staged in registers across the WMMA section.
template<bool ABF, bool OBF>
__global__ __launch_bounds__(256) void gemm_kernel(const void* __restrict__ Aptr,
                                                   const float* __restrict__ W,
                                                   void* __restrict__ Cptr,
                                                   int M, int N, int K) {
  (void)M;
  __shared__ __bf16 lA[2][128 * 72];  // 64 k + 8 pad -> conflict-free b128 reads
  __shared__ __bf16 lW[2][128 * 72];
  const int t    = threadIdx.x;
  const int lane = t & 31;
  const int wid  = t >> 5;
  const int wm   = wid & 3;        // 4 waves along M
  const int wn   = wid >> 2;       // 2 waves along N
  const int m    = lane & 15;
  const int half = lane >> 4;
  const int lrow = t >> 1;         // 0..127
  const int lcol = (t & 1) * 32;   // 0 or 32

  const float*  Af = (const float*)Aptr;
  const __bf16* Ab = (const __bf16*)Aptr;
  const size_t arow = (size_t)(blockIdx.y * 128 + lrow) * (size_t)K;
  const size_t wrow = (size_t)(blockIdx.x * 128 + lrow) * (size_t)K;

  v8f acc[2][4];
  for (int i = 0; i < 2; i++)
    for (int j = 0; j < 4; j++)
      for (int e = 0; e < 8; e++) acc[i][j][e] = 0.0f;

  // ---- prologue: stage tile 0 into buffer 0 ----
  {
    uint4* dstA = (uint4*)&lA[0][lrow * 72 + lcol];
    if (ABF) {
      const uint4* src = (const uint4*)(Ab + arow + lcol);
      dstA[0] = src[0]; dstA[1] = src[1]; dstA[2] = src[2]; dstA[3] = src[3];
    } else {
      const float4* src = (const float4*)(Af + arow + lcol);
      for (int i = 0; i < 4; i++) dstA[i] = pk16(src[2 * i], src[2 * i + 1]);
    }
    const float4* srcW = (const float4*)(W + wrow + lcol);
    uint4* dstW = (uint4*)&lW[0][lrow * 72 + lcol];
    for (int i = 0; i < 4; i++) dstW[i] = pk16(srcW[2 * i], srcW[2 * i + 1]);
  }
  __syncthreads();

  int cur = 0;
  for (int k0 = 0; k0 < K; k0 += 64) {
    const int kn = k0 + 64;
    const bool more = kn < K;

    // ---- issue next tile's global loads (hidden under WMMAs below) ----
    float4 ra[8], rw[8];
    uint4  rau[4];
    if (more) {
      if (ABF) {
        const uint4* s = (const uint4*)(Ab + arow + kn + lcol);
        rau[0] = s[0]; rau[1] = s[1]; rau[2] = s[2]; rau[3] = s[3];
      } else {
        const float4* s = (const float4*)(Af + arow + kn + lcol);
        for (int i = 0; i < 8; i++) ra[i] = s[i];
      }
      const float4* sw = (const float4*)(W + wrow + kn + lcol);
      for (int i = 0; i < 8; i++) rw[i] = sw[i];
      if (kn + 64 < K) {  // gfx1250 global_prefetch_b8, two tiles ahead
        __builtin_prefetch(ABF ? (const void*)(Ab + arow + kn + 64 + lcol)
                               : (const void*)(Af + arow + kn + 64 + lcol), 0, 1);
        __builtin_prefetch((const void*)(W + wrow + kn + 64 + lcol), 0, 1);
      }
    }

    // ---- compute from current buffer ----
    for (int ks = 0; ks < 64; ks += 32) {
      v16bf af[2], wb[4];
      for (int mt = 0; mt < 2; mt++) {
        const __bf16* p = &lA[cur][(32 * wm + 16 * mt + m) * 72 + ks + 8 * half];
        af[mt] = mkfrag(p, p + 16);       // A: k = 8h..8h+7 and 16+8h..16+8h+7
      }
      for (int nt = 0; nt < 4; nt++) {
        const __bf16* p = &lW[cur][(64 * wn + 16 * nt + m) * 72 + ks + 8 * half];
        wb[nt] = mkfrag(p, p + 16);       // B: k = 8h..8h+7 and 16+8h..16+8h+7
      }
      for (int mt = 0; mt < 2; mt++)
        for (int nt = 0; nt < 4; nt++)
          acc[mt][nt] = __builtin_amdgcn_wmma_f32_16x16x32_bf16(
              false, af[mt], false, wb[nt], (short)0, acc[mt][nt], false, false);
    }

    // ---- store staged tile to the other buffer ----
    if (more) {
      uint4* dstA = (uint4*)&lA[cur ^ 1][lrow * 72 + lcol];
      if (ABF) {
        dstA[0] = rau[0]; dstA[1] = rau[1]; dstA[2] = rau[2]; dstA[3] = rau[3];
      } else {
        for (int i = 0; i < 4; i++) dstA[i] = pk16(ra[2 * i], ra[2 * i + 1]);
      }
      uint4* dstW = (uint4*)&lW[cur ^ 1][lrow * 72 + lcol];
      for (int i = 0; i < 4; i++) dstW[i] = pk16(rw[2 * i], rw[2 * i + 1]);
    }
    __syncthreads();
    cur ^= 1;
  }

  // ---- epilogue ----
  for (int mt = 0; mt < 2; mt++) {
    for (int nt = 0; nt < 4; nt++) {
      const int col = blockIdx.x * 128 + 64 * wn + 16 * nt + m;
      for (int r = 0; r < 8; r++) {
        const int row = blockIdx.y * 128 + 32 * wm + 16 * mt + 8 * half + r;
        const float vv = acc[mt][nt][r];
        if (OBF) ((__bf16*)Cptr)[(size_t)row * N + col] = f2bf(vv);
        else     ((float*)Cptr)[(size_t)row * N + col]  = vv;
      }
    }
  }
}

// ---------------- RoPE (in place, bf16 storage, f32 math) ----------------
// `scale` folds the attention 1/sqrt(HD) into Q at full f32 precision.
__global__ void rope_kernel(__bf16* __restrict__ X, int nh, float scale) {
  const int t = blockIdx.x * 256 + threadIdx.x;
  const int d = t & 63;
  const int h = (t >> 6) % nh;
  const int s = (t / (64 * nh)) % S_LEN;
  const int b = t / (64 * nh * S_LEN);
  const float inv = __powf(10000.0f, -(float)(2 * d) * (1.0f / 128.0f));
  const float ang = (float)s * inv;
  const float c = cosf(ang), sn = sinf(ang);
  const size_t base = (((size_t)(b * S_LEN + s)) * nh + h) * HD + d;
  const float x1 = bf2f(X[base]);
  const float x2 = bf2f(X[base + 64]);
  X[base]      = f2bf((x1 * c - x2 * sn) * scale);
  X[base + 64] = f2bf((x2 * c + x1 * sn) * scale);
}

// ---------------- V transpose: Vp[b][s][kvh][d] -> Vt[b][kvh][d][s] ----------
__global__ void vtrans_kernel(const __bf16* __restrict__ Vp, __bf16* __restrict__ Vt) {
  __shared__ __bf16 tile[32][34];
  const int s0 = blockIdx.x * 32, d0 = blockIdx.y * 32;
  const int z = blockIdx.z, b = z >> 3, kvh = z & 7;
  const int tx = threadIdx.x, ty = threadIdx.y;
  for (int i = 0; i < 4; i++) {
    const int s = s0 + ty + i * 8;
    tile[ty + i * 8][tx] = Vp[(((size_t)(b * S_LEN + s)) * KVH + kvh) * HD + d0 + tx];
  }
  __syncthreads();
  for (int i = 0; i < 4; i++) {
    const int d = d0 + ty + i * 8;
    Vt[((size_t)((b * KVH + kvh) * HD + d)) * S_LEN + s0 + tx] = tile[tx][ty + i * 8];
  }
}

// ---------------- Flash attention (causal, GQA 4:1) ----------------
// grid = (S/128, HEADS, B), block = 256. Each wave: 16 queries x 128 dims.
// Q comes in pre-scaled by 1/sqrt(HD). K/V stream from 192MB L2 (bf16 K+V =
// 16.8MB total). K frags batched before score WMMAs; V frags issued before
// the softmax VALU so exp/shuffle work hides their latency.
__global__ __launch_bounds__(256) void attn_kernel(const __bf16* __restrict__ Qb,
                                                   const __bf16* __restrict__ Kb,
                                                   const __bf16* __restrict__ Vt,
                                                   __bf16* __restrict__ Ctx) {
  __shared__ __bf16 lP[8][16 * 40];  // per-wave P scratch, padded rows
  const int lane = threadIdx.x & 31;
  const int wid  = threadIdx.x >> 5;
  const int m    = lane & 15;
  const int half = lane >> 4;
  const int b = blockIdx.z, h = blockIdx.y;
  const int kvh = h >> 2;                  // 32 heads -> 8 kv heads
  const int s0 = blockIdx.x * 128 + wid * 16;

  // Q fragments: 4 chunks of 32 dims
  v16bf qa[4];
  {
    const __bf16* qr = Qb + (((size_t)(b * S_LEN + s0 + m)) * HEADS + h) * HD;
    for (int c = 0; c < 4; c++)
      qa[c] = mkfrag(qr + c * 32 + 8 * half, qr + c * 32 + 16 + 8 * half);
  }

  v8f acc[8];
  for (int nt = 0; nt < 8; nt++)
    for (int e = 0; e < 8; e++) acc[nt][e] = 0.0f;
  float m_st[8], l_st[8];
  for (int r = 0; r < 8; r++) { m_st[r] = -1e30f; l_st[r] = 0.0f; }

  const __bf16* kbase0 = Kb + ((size_t)b * S_LEN * KVH + kvh) * HD;
  const __bf16* vbase  = Vt + ((size_t)((b * KVH + kvh) * HD + m)) * S_LEN;
  const int nb = (s0 + 47) >> 5;           // causal block count (32 keys each)

  for (int j = 0; j < nb; j++) {
    const int k0 = j * 32;

    // ---- batch all 8 K fragments, then 8 score WMMAs ----
    v16bf kf[8];
    for (int c = 0; c < 4; c++)
      for (int t2 = 0; t2 < 2; t2++) {
        const __bf16* kr = kbase0 + (size_t)(k0 + 16 * t2 + m) * (KVH * HD)
                                  + c * 32 + 8 * half;
        kf[c * 2 + t2] = mkfrag(kr, kr + 16);
      }
    v8f sc[2];
    for (int t2 = 0; t2 < 2; t2++)
      for (int e = 0; e < 8; e++) sc[t2][e] = 0.0f;
    for (int c = 0; c < 4; c++) {
      sc[0] = __builtin_amdgcn_wmma_f32_16x16x32_bf16(
          false, qa[c], false, kf[c * 2 + 0], (short)0, sc[0], false, false);
      sc[1] = __builtin_amdgcn_wmma_f32_16x16x32_bf16(
          false, qa[c], false, kf[c * 2 + 1], (short)0, sc[1], false, false);
    }

    // ---- issue V fragment loads now; softmax VALU below hides them ----
    v16bf vf[8];
    for (int nt = 0; nt < 8; nt++) {
      const __bf16* vr = vbase + (size_t)(nt * 16) * S_LEN + k0 + 8 * half;
      vf[nt] = mkfrag(vr, vr + 16);
    }

    // ---- causal mask only on diagonal blocks (wave-uniform branch) ----
    if (k0 + 31 > s0) {
      for (int t2 = 0; t2 < 2; t2++) {
        const int key = k0 + 16 * t2 + m;
        for (int r = 0; r < 8; r++) {
          const int qi = s0 + r + 8 * half;
          if (key > qi) sc[t2][r] = -1e30f;
        }
      }
    }

    // ---- online softmax per row (16-lane butterflies within half-wave) ----
    float cr[8];
    for (int r = 0; r < 8; r++) {
      float bm = fmaxf(sc[0][r], sc[1][r]);
      bm = fmaxf(bm, __shfl_xor(bm, 1, 32));
      bm = fmaxf(bm, __shfl_xor(bm, 2, 32));
      bm = fmaxf(bm, __shfl_xor(bm, 4, 32));
      bm = fmaxf(bm, __shfl_xor(bm, 8, 32));
      const float mn = fmaxf(m_st[r], bm);
      const float corr = __expf(m_st[r] - mn);
      const float p0 = __expf(sc[0][r] - mn);
      const float p1 = __expf(sc[1][r] - mn);
      float rs = p0 + p1;
      rs += __shfl_xor(rs, 1, 32);
      rs += __shfl_xor(rs, 2, 32);
      rs += __shfl_xor(rs, 4, 32);
      rs += __shfl_xor(rs, 8, 32);
      l_st[r] = l_st[r] * corr + rs;
      m_st[r] = mn;
      cr[r] = corr;
      // P tile to per-wave LDS (C-layout -> memory row-major)
      lP[wid][(r + 8 * half) * 40 + m]      = f2bf(p0);
      lP[wid][(r + 8 * half) * 40 + 16 + m] = f2bf(p1);
    }
    for (int nt = 0; nt < 8; nt++)
      for (int r = 0; r < 8; r++) acc[nt][r] *= cr[r];

    // wave-private LDS RAW: ensure all ds stores landed before frag reads
    asm volatile("s_wait_dscnt 0x0" ::: "memory");
    v16bf pf = mkfrag(&lP[wid][m * 40 + 8 * half], &lP[wid][m * 40 + 8 * half + 16]);

    // ---- ctx += P * V ----
    for (int nt = 0; nt < 8; nt++)
      acc[nt] = __builtin_amdgcn_wmma_f32_16x16x32_bf16(
          false, pf, false, vf[nt], (short)0, acc[nt], false, false);
  }

  for (int r = 0; r < 8; r++) l_st[r] = 1.0f / l_st[r];
  for (int nt = 0; nt < 8; nt++) {
    for (int r = 0; r < 8; r++) {
      const int row = s0 + r + 8 * half;
      const int col = nt * 16 + m;
      Ctx[(((size_t)(b * S_LEN + row)) * HEADS + h) * HD + col] = f2bf(acc[nt][r] * l_st[r]);
    }
  }
}

// ---------------- orchestration ----------------
extern "C" void kernel_launch(void* const* d_in, const int* in_sizes, int n_in,
                              void* d_out, int out_size, void* d_ws, size_t ws_size,
                              hipStream_t stream) {
  (void)in_sizes; (void)n_in; (void)out_size; (void)ws_size;
  const float* q  = (const float*)d_in[0];
  const float* k  = (const float*)d_in[1];
  const float* v  = (const float*)d_in[2];
  const float* Wq = (const float*)d_in[3];
  const float* Wk = (const float*)d_in[4];
  const float* Wv = (const float*)d_in[5];
  const float* Wd = (const float*)d_in[6];
  float* out = (float*)d_out;

  char* ws = (char*)d_ws;
  __bf16* Qb  = (__bf16*)(ws);                 // 32 MB  [b][s][h][d]
  __bf16* Kb  = (__bf16*)(ws + 33554432);      //  8 MB  [b][s][kvh][d]
  __bf16* Vp  = (__bf16*)(ws + 41943040);      //  8 MB  [b][s][kvh][d]
  __bf16* Vt  = (__bf16*)(ws + 50331648);      //  8 MB  [b][kvh][d][s]
  __bf16* Ctx = (__bf16*)(ws + 58720256);      // 32 MB  [b][s][h][d]

  const dim3 blk(256);
  // projections: M = B*S = 4096, K = 4096
  gemm_kernel<false, true><<<dim3(32, 32), blk, 0, stream>>>(q, Wq, Qb, 4096, 4096, 4096);
  gemm_kernel<false, true><<<dim3(8, 32),  blk, 0, stream>>>(k, Wk, Kb, 4096, 1024, 4096);
  gemm_kernel<false, true><<<dim3(8, 32),  blk, 0, stream>>>(v, Wv, Vp, 4096, 1024, 4096);
  // RoPE on Q (pre-scaled by 1/sqrt(HD)) and K
  rope_kernel<<<32768, 256, 0, stream>>>(Qb, HEADS, 0.08838834764831845f);
  rope_kernel<<<8192, 256, 0, stream>>>(Kb, KVH, 1.0f);
  // V transpose for PV B-fragments
  vtrans_kernel<<<dim3(64, 4, 16), dim3(32, 8), 0, stream>>>(Vp, Vt);
  // causal flash attention
  attn_kernel<<<dim3(16, 32, 2), blk, 0, stream>>>(Qb, Kb, Vt, Ctx);
  // output projection (f32 out)
  gemm_kernel<true, false><<<dim3(32, 32), blk, 0, stream>>>(Ctx, Wd, out, 4096, 4096, 4096);
}
